// RNN_20409684590856
// MI455X (gfx1250) — matrix operand order repaired and checked by reading, not compiled, not run on hardware
//
#include <hip/hip_runtime.h>
#include <hip/hip_bf16.h>

#define B_ 64
#define T_ 2048
#define I_ 128
#define H_ 256
#define HP_ 264   // padded LDS row stride in halfs: 528 B -> 16 B bank phase per
                  // row, so lanes 0..15 (rows) spread across all 64 LDS banks

typedef __attribute__((ext_vector_type(16))) _Float16 v16h;
typedef __attribute__((ext_vector_type(8)))  _Float16 v8h;
typedef __attribute__((ext_vector_type(8)))  float    v8f;

union Frag16 {
    v16h v;
    v8h  h[2];
    _Float16 e[16];
};

// ---------------------------------------------------------------------------
// Phase 1: xproj[t][b][n] = sum_k x[b][t][k] * Wx[k][n] + bias[n]
// One wave per 16x16 output tile. M dimension = T*B rows (row r: t=r/64, b=r%64,
// so a 16-row tile sits inside a single t). K = I = 128 -> 4 WMMA k-steps.
// ---------------------------------------------------------------------------
__global__ void __launch_bounds__(256)
rnn_xproj_kernel(const float* __restrict__ x, const float* __restrict__ W,
                 const float* __restrict__ bias, float* __restrict__ xproj)
{
    const int lane  = threadIdx.x & 31;
    const int wave  = threadIdx.x >> 5;
    const int tile  = blockIdx.x * 8 + wave;     // 131072 tiles total
    const int mtile = tile >> 4;                  // 0..8191 (16-row strip of T*B)
    const int ntile = tile & 15;                  // 0..15   (16-col strip of H)
    const int lhalf = lane >> 4;                  // half-wave select
    const int lmod  = lane & 15;

    // A operand: this lane owns row M = lmod of the tile.
    const int r  = mtile * 16 + lmod;             // row in (T*B)
    const int t  = r >> 6;
    const int bb = r & 63;
    const float* xrow = x + ((size_t)bb * T_ + t) * I_;   // x is (B,T,I)

    v8f acc0 = {}, acc1 = {};
#pragma unroll
    for (int s = 0; s < 4; ++s) {
        // A 16x32 f16: lane half selects K octet; elements 0-7 -> K = k0..k0+7,
        // elements 8-15 -> K = k0+16..k0+23   (ISA 7.12.2, 16-bit A layout)
        Frag16 a;
        const int k0 = s * 32 + lhalf * 8;
#pragma unroll
        for (int e = 0; e < 8; ++e) a.e[e]     = (_Float16)xrow[k0 + e];
#pragma unroll
        for (int e = 0; e < 8; ++e) a.e[8 + e] = (_Float16)xrow[k0 + 16 + e];

        // B 32x16 f16: lane column N = lmod, elements follow the same K order as A.
        Frag16 bf;
        const float* wrow = W + (size_t)(s * 32 + lhalf * 8) * H_ + ntile * 16 + lmod;
#pragma unroll
        for (int e = 0; e < 16; ++e) bf.e[e] = (_Float16)wrow[(size_t)(e + (e >> 3) * 8) * H_];

        if (s & 1)
            acc1 = __builtin_amdgcn_wmma_f32_16x16x32_f16(
                false, a.v, false, bf.v, (short)0, acc1, false, false);
        else
            acc0 = __builtin_amdgcn_wmma_f32_16x16x32_f16(
                false, a.v, false, bf.v, (short)0, acc0, false, false);
    }

    // C/D layout: element v -> row M = v + 8*lhalf, col N = lmod.
    const int n  = ntile * 16 + lmod;
    const float bn = bias[n];
#pragma unroll
    for (int v = 0; v < 8; ++v) {
        const int m  = v + 8 * lhalf;
        const int rr = mtile * 16 + m;            // rr = t*64 + b
        xproj[(size_t)rr * H_ + n] = acc0[v] + acc1[v] + bn;  // xproj[t][b][n]
    }
}

// ---------------------------------------------------------------------------
// Phase 2: sequential scan. 4 workgroups (one per 16-row batch strip), each
// 512 threads = 16 waves; wave w owns output columns [16w, 16w+16).
// Wh fragments (8 ksteps x v16h = 64 VGPRs/lane) are register-resident for
// the whole scan. h (16 x H, padded stride HP_) is double-buffered in LDS;
// padding makes the per-kstep ds_load_b128 A-loads bank-conflict-free.
// Four accumulators keep the XDL dependency chains at depth 2; xproj reads
// for step t+1 are issued at the top of step t to hide L2 latency.
// One __syncthreads per timestep.
// ---------------------------------------------------------------------------
__global__ void __launch_bounds__(512)
rnn_scan_kernel(const float* __restrict__ W, const float* __restrict__ xproj,
                float* __restrict__ out)
{
    __shared__ _Float16 hbuf[2][16][HP_];         // ~16.5 KB double buffer

    const int tid   = threadIdx.x;
    const int lane  = tid & 31;
    const int wave  = tid >> 5;                   // 0..15 = output ntile
    const int lhalf = lane >> 4;
    const int lmod  = lane & 15;
    const int b0    = blockIdx.x * 16;            // batch strip base

    // Preload this wave's Wh B-fragments into registers (once).
    // B 32x16 f16: lane column N = lmod, elements follow the same K order as A.
    Frag16 bfrag[8];
#pragma unroll
    for (int s = 0; s < 8; ++s) {
        const float* wrow = W + (size_t)(I_ + s * 32 + lhalf * 8) * H_ + wave * 16 + lmod;
#pragma unroll
        for (int e = 0; e < 16; ++e) bfrag[s].e[e] = (_Float16)wrow[(size_t)(e + (e >> 3) * 8) * H_];
    }

    // h0 = 0 (zero both buffers' payload region)
    for (int i = tid; i < 16 * HP_; i += 512) {
        hbuf[0][i / HP_][i % HP_] = (_Float16)0.0f;
        hbuf[1][i / HP_][i % HP_] = (_Float16)0.0f;
    }
    __syncthreads();

    const int n = wave * 16 + lmod;               // this lane's output column

    // Software pipeline: xpv holds xproj[t] for the current step.
    float xpv[8];
#pragma unroll
    for (int v = 0; v < 8; ++v) {
        const int m = v + 8 * lhalf;
        xpv[v] = xproj[(size_t)(b0 + m) * H_ + n];           // t = 0
    }

    for (int t = 0; t < T_; ++t) {
        // Issue next step's xproj loads first so they overlap the WMMA chain
        // and survive across the barrier (full step of latency hiding).
        const int tn = (t + 1 < T_) ? (t + 1) : t;
        const float* xq = xproj + (size_t)tn * (B_ * H_);
        float xpn[8];
#pragma unroll
        for (int v = 0; v < 8; ++v) {
            const int m = v + 8 * lhalf;
            xpn[v] = xq[(size_t)(b0 + m) * H_ + n];
        }

        const _Float16* hc = &hbuf[t & 1][0][0];
        _Float16*       hn = &hbuf[(t + 1) & 1][0][0];

        v8f acc[4] = {v8f{}, v8f{}, v8f{}, v8f{}};
#pragma unroll
        for (int s = 0; s < 8; ++s) {
            // A 16x32 from row-major padded h: two 16B LDS loads per lane,
            // conflict-free thanks to the 528B row stride.
            Frag16 a;
            const int k0 = s * 32 + lhalf * 8;
            a.h[0] = *(const v8h*)(hc + lmod * HP_ + k0);
            a.h[1] = *(const v8h*)(hc + lmod * HP_ + k0 + 16);
            acc[s & 3] = __builtin_amdgcn_wmma_f32_16x16x32_f16(
                false, a.v, false, bfrag[s].v, (short)0, acc[s & 3], false, false);
        }

        float* op = out + (size_t)t * (B_ * H_);
        float hvals[8];
#pragma unroll
        for (int v = 0; v < 8; ++v) {
            const int m = v + 8 * lhalf;          // row within strip
            const float val  = (acc[0][v] + acc[1][v]) + (acc[2][v] + acc[3][v])
                               + xpv[v];
            const float hval = 1.0f / (1.0f + __expf(-val));
            hvals[v] = hval;
            // Critical path: LDS write feeds the next timestep.
            hn[m * HP_ + n] = (_Float16)hval;
            xpv[v] = xpn[v];
        }
#pragma unroll
        for (int v = 0; v < 8; ++v) {
            const int m = v + 8 * lhalf;
            // out is write-once: non-temporal keeps L2 free for xproj reuse.
            __builtin_nontemporal_store(hvals[v], &op[(size_t)(b0 + m) * H_ + n]);
        }
        __syncthreads();   // hn complete before next step reads it; also fences
                           // reuse of hc's buffer at step t+1 (double buffered)
    }
}

// ---------------------------------------------------------------------------
extern "C" void kernel_launch(void* const* d_in, const int* in_sizes, int n_in,
                              void* d_out, int out_size, void* d_ws, size_t ws_size,
                              hipStream_t stream)
{
    const float* x    = (const float*)d_in[0];    // (B,T,I) f32
    const float* W    = (const float*)d_in[1];    // (I+H,H) f32
    const float* bias = (const float*)d_in[2];    // (H,)    f32
    float* out   = (float*)d_out;                 // (T,B,H) f32
    float* xproj = (float*)d_ws;                  // (T,B,H) f32 scratch, 128 MiB

    // Phase 1: big parallel GEMM, fills the whole GPU.
    // tiles = (T*B/16) * (H/16) = 131072 waves -> 16384 blocks of 8 waves.
    rnn_xproj_kernel<<<dim3(16384), dim3(256), 0, stream>>>(x, W, bias, xproj);

    // Phase 2: 4 independent scan chains (one WGP each), register-resident Wh.
    rnn_scan_kernel<<<dim3(4), dim3(512), 0, stream>>>(W, xproj, out);
}
